// _SequenceModel_89309549953279
// MI455X (gfx1250) — matrix operand order repaired and checked by reading, not compiled, not run on hardware
//
#include <hip/hip_runtime.h>
#include <hip/hip_bf16.h>

// ---------------------------------------------------------------------------
// 2-layer LSTM (H=512, B=1024) encode + autoregressive decode on CDNA5.
// Core op: gates = [h...] @ W^T via v_wmma_f32_16x16x32_bf16, fused cell.
// Each wave now owns TWO 16-row m-tiles so every L2-sourced B fragment feeds
// two WMMAs (halves L2 weight traffic; the recurrent chain is L2-bound).
// ---------------------------------------------------------------------------

#define H_DIM   512
#define B_DIM   1024
#define G4      (4 * H_DIM)   // 2048

typedef __attribute__((ext_vector_type(16))) __bf16 v16bf;
typedef __attribute__((ext_vector_type(8)))  float  v8f;

union FragB { uint4 u[2]; v16bf v; };   // 16 bf16 = 32 B
union FragF { v8f v; float f[8]; };     // 8 f32 accumulator

__device__ __forceinline__ float sigm_(float x) {
    return 1.0f / (1.0f + __expf(-x));
}
__device__ __forceinline__ float tanh_(float x) {
    // 1 - 2/(exp(2x)+1): saturates cleanly to +/-1, no NaN at large |x|
    return 1.0f - 2.0f / (__expf(2.0f * x) + 1.0f);
}

// ---------------------------------------------------------------------------
// One LSTM time step for one layer.
//   gates[b, n] = sum_k A[b,k] * W[n,k]  (+ bias[n])  (+ xval[b]*xw[n])
//   A rows are the concat of A0 (k<512) and optional A1 (k>=512).
// Grid: (4 j-groups of 128 cols, 32 m-pair-tiles of 32 rows). Block: 256.
// Each wave: one 16-col j-tile x TWO 16-row m-tiles x ALL FOUR gates.
// K-loop: fully unrolled, one-stage software pipeline; per stage 8 WMMAs
// cover the next stage's 8 global + 4 ds loads.
// ---------------------------------------------------------------------------
template <int K, bool HASX>
__global__ void __launch_bounds__(256)
lstm_step_kernel(const __hip_bfloat16* __restrict__ A0,   // [1024][512] bf16
                 const __hip_bfloat16* __restrict__ A1,   // [1024][512] bf16 or null
                 const __hip_bfloat16* __restrict__ W,    // [2048][K] bf16 row-major
                 const float* __restrict__ bias,          // [2048]
                 const float* __restrict__ xsrc,          // xval[m] = xsrc[m*x_stride]
                 int x_stride,
                 const float* __restrict__ xw,            // [2048] f32 (W_ih_0)
                 float* __restrict__ cstate,              // [1024][512] f32 (in-place)
                 __hip_bfloat16* __restrict__ h_out)      // [1024][512] bf16
{
    constexpr int LDK   = K + 8;     // LDS row pitch (bf16 elems), bank-skewed
    constexpr int NSTEP = K / 32;
    constexpr size_t GATE_STRIDE = (size_t)H_DIM * K * 2;   // bytes between gates

    extern __shared__ char smem[];
    __hip_bfloat16* lds_a = (__hip_bfloat16*)smem;                      // [32][LDK]
    float*          lds_x = (float*)(smem + (size_t)32 * LDK * 2);      // [32]

    const int tid = threadIdx.x;
    const int m0  = blockIdx.y * 32;

    // ---- stage A tile [32 x K] into LDS (16-byte chunks) ----
    constexpr int CPR = K >> 3;                   // 8-bf16 chunks per row
#pragma unroll 4
    for (int idx = tid; idx < 32 * CPR; idx += 256) {
        int row = idx / CPR;
        int k   = (idx - row * CPR) << 3;
        const __hip_bfloat16* src =
            (k < H_DIM) ? (A0 + (size_t)(m0 + row) * H_DIM + k)
                        : (A1 + (size_t)(m0 + row) * H_DIM + (k - H_DIM));
        *(uint4*)(lds_a + (size_t)row * LDK + k) = *(const uint4*)src;
    }
    if (HASX && tid < 32)
        lds_x[tid] = xsrc[(size_t)(m0 + tid) * x_stride];
    __syncthreads();

    // ---- per-wave tile ----
    const int wave = tid >> 5;
    const int lane = tid & 31;
    const int j0   = blockIdx.x * 128 + wave * 16;        // j column base (0..511)
    const int row  = lane & 15;                           // A row within tile
    const int aco  = (lane >> 4) * 16;                    // A K sub-offset (contiguous, matches B)
    const int bko  = (lane >> 4) * 16;                    // B K sub-offset
    const int ncol = j0 + (lane & 15);                    // j column for B / epilogue

    // Single base pointers; per-step / per-gate offsets fold into immediates.
    const __hip_bfloat16* arow0 = lds_a + (size_t)row * LDK + aco;          // m-tile 0
    const __hip_bfloat16* arow1 = arow0 + (size_t)16 * LDK;                 // m-tile 1
    const char*           wbase = (const char*)W + ((size_t)ncol * K + bko) * 2;

    FragF acc[2][4];
#pragma unroll
    for (int mt = 0; mt < 2; ++mt)
#pragma unroll
        for (int g = 0; g < 4; ++g)
#pragma unroll
            for (int e = 0; e < 8; ++e) acc[mt][g].f[e] = 0.0f;

    // ---- pipelined K loop (fully unrolled) ----
    FragB ca0, ca1, cb[4];
    ca0.u[0] = *(const uint4*)(arow0);
    ca0.u[1] = *(const uint4*)(arow0 + 8);
    ca1.u[0] = *(const uint4*)(arow1);
    ca1.u[1] = *(const uint4*)(arow1 + 8);
#pragma unroll
    for (int g = 0; g < 4; ++g) {
        const uint4* p = (const uint4*)(wbase + (size_t)g * GATE_STRIDE);
        cb[g].u[0] = p[0];
        cb[g].u[1] = p[1];
    }

#pragma unroll
    for (int s = 0; s < NSTEP; ++s) {
        FragB na0, na1, nb[4];
        if (s + 1 < NSTEP) {
            const int kn = (s + 1) * 32;
            na0.u[0] = *(const uint4*)(arow0 + kn);
            na0.u[1] = *(const uint4*)(arow0 + kn + 8);
            na1.u[0] = *(const uint4*)(arow1 + kn);
            na1.u[1] = *(const uint4*)(arow1 + kn + 8);
#pragma unroll
            for (int g = 0; g < 4; ++g) {
                const uint4* p = (const uint4*)(wbase + (size_t)g * GATE_STRIDE
                                                      + (size_t)kn * 2);
                nb[g].u[0] = p[0];
                nb[g].u[1] = p[1];
            }
        }
        // 8 independent WMMAs: each B fragment feeds both m-tiles.
#pragma unroll
        for (int g = 0; g < 4; ++g) {
            acc[0][g].v = __builtin_amdgcn_wmma_f32_16x16x32_bf16(
                false, ca0.v, false, cb[g].v, (short)0, acc[0][g].v, false, false);
            acc[1][g].v = __builtin_amdgcn_wmma_f32_16x16x32_bf16(
                false, ca1.v, false, cb[g].v, (short)0, acc[1][g].v, false, false);
        }
        if (s + 1 < NSTEP) {
            ca0 = na0;
            ca1 = na1;
#pragma unroll
            for (int g = 0; g < 4; ++g) cb[g] = nb[g];
        }
    }

    // ---- fused LSTM cell epilogue (both m-tiles) ----
    float bi[4], xwv[4];
#pragma unroll
    for (int g = 0; g < 4; ++g) {
        bi[g]  = bias[g * H_DIM + ncol];
        xwv[g] = HASX ? xw[g * H_DIM + ncol] : 0.0f;
    }
    const int mh = (lane >> 4) * 8;
#pragma unroll
    for (int mt = 0; mt < 2; ++mt) {
#pragma unroll
        for (int r = 0; r < 8; ++r) {
            int   ml = 16 * mt + r + mh;          // C layout: M = r + 8*(lane/16)
            int   m  = m0 + ml;
            float xv = HASX ? lds_x[ml] : 0.0f;
            float gi = acc[mt][0].f[r] + bi[0] + xv * xwv[0];
            float gf = acc[mt][1].f[r] + bi[1] + xv * xwv[1];
            float gg = acc[mt][2].f[r] + bi[2] + xv * xwv[2];
            float go = acc[mt][3].f[r] + bi[3] + xv * xwv[3];
            float it = sigm_(gi), ft = sigm_(gf), gt = tanh_(gg), ot = sigm_(go);
            size_t idx = (size_t)m * H_DIM + ncol;
            float  cn  = ft * cstate[idx] + it * gt;
            cstate[idx] = cn;
            h_out[idx]  = __float2bfloat16(ot * tanh_(cn));
        }
    }
}

// ---------------------------------------------------------------------------
// Head: pred[b] = h1[b,:] . head_w + head_b ; also writes out[b, t].
// One wave32 per batch row.
// ---------------------------------------------------------------------------
__global__ void __launch_bounds__(256)
head_kernel(const __hip_bfloat16* __restrict__ h1, const float* __restrict__ head_w,
            const float* __restrict__ head_b, float* __restrict__ pred,
            float* __restrict__ out, int t, int horizon)
{
    int b    = blockIdx.x * 8 + (threadIdx.x >> 5);
    int lane = threadIdx.x & 31;
    float s = 0.0f;
#pragma unroll
    for (int k = lane; k < H_DIM; k += 32)
        s += __bfloat162float(h1[(size_t)b * H_DIM + k]) * head_w[k];
#pragma unroll
    for (int off = 16; off > 0; off >>= 1)
        s += __shfl_down(s, off, 32);
    if (lane == 0) {
        float p = s + head_b[0];
        pred[b] = p;
        out[(size_t)b * horizon + t] = p;
    }
}

// ---------------------------------------------------------------------------
// Utility kernels
// ---------------------------------------------------------------------------
__global__ void f32_to_bf16_kernel(const float* __restrict__ s,
                                   __hip_bfloat16* __restrict__ d, int n)
{
    int i = blockIdx.x * 256 + threadIdx.x;
    if (i < n) d[i] = __float2bfloat16(s[i]);
}

// Wcat[n][k] = k<512 ? W_ih_1[n][k] : W_hh_1[n][k-512]   ([2048][1024] bf16)
__global__ void build_wcat_kernel(const float* __restrict__ wih,
                                  const float* __restrict__ whh,
                                  __hip_bfloat16* __restrict__ d)
{
    int i = blockIdx.x * 256 + threadIdx.x;
    if (i < G4 * 1024) {
        int n = i >> 10, k = i & 1023;
        float v = (k < H_DIM) ? wih[(size_t)n * H_DIM + k]
                              : whh[(size_t)n * H_DIM + (k - H_DIM)];
        d[i] = __float2bfloat16(v);
    }
}

__global__ void zero_bytes_kernel(uint4* __restrict__ p, size_t n16)
{
    size_t i = (size_t)blockIdx.x * 256 + threadIdx.x;
    size_t stride = (size_t)gridDim.x * 256;
    uint4 z = {0u, 0u, 0u, 0u};
    for (; i < n16; i += stride) p[i] = z;
}

// ---------------------------------------------------------------------------
// Host driver
// ---------------------------------------------------------------------------
extern "C" void kernel_launch(void* const* d_in, const int* in_sizes, int n_in,
                              void* d_out, int out_size, void* d_ws, size_t ws_size,
                              hipStream_t stream)
{
    (void)n_in; (void)ws_size;
    const float* x     = (const float*)d_in[0];   // [1024][T] row-major
    const float* Wih0  = (const float*)d_in[1];   // [2048][1] -> [2048]
    const float* Whh0  = (const float*)d_in[2];   // [2048][512]
    const float* b0    = (const float*)d_in[3];   // [2048]
    const float* Wih1  = (const float*)d_in[4];   // [2048][512]
    const float* Whh1  = (const float*)d_in[5];   // [2048][512]
    const float* b1    = (const float*)d_in[6];   // [2048]
    const float* headw = (const float*)d_in[7];   // [512]
    const float* headb = (const float*)d_in[8];   // [1]

    const int T       = in_sizes[0] / B_DIM;            // 256
    const int horizon = out_size / B_DIM;               // 64
    float* out = (float*)d_out;

    // ---- workspace layout ----
    char* ws = (char*)d_ws;
    const size_t SZ_WHH0 = (size_t)G4 * H_DIM * 2;      // 2 MB
    const size_t SZ_WCAT = (size_t)G4 * 1024  * 2;      // 4 MB
    const size_t SZ_H    = (size_t)B_DIM * H_DIM * 2;   // 1 MB
    const size_t SZ_C    = (size_t)B_DIM * H_DIM * 4;   // 2 MB
    size_t off = 0;
    __hip_bfloat16* Whh0b = (__hip_bfloat16*)(ws + off); off += SZ_WHH0;
    __hip_bfloat16* Wcat  = (__hip_bfloat16*)(ws + off); off += SZ_WCAT;
    size_t state_off = off;
    __hip_bfloat16* h0b[2], *h1b[2];
    h0b[0] = (__hip_bfloat16*)(ws + off); off += SZ_H;
    h0b[1] = (__hip_bfloat16*)(ws + off); off += SZ_H;
    h1b[0] = (__hip_bfloat16*)(ws + off); off += SZ_H;
    h1b[1] = (__hip_bfloat16*)(ws + off); off += SZ_H;
    float* c0   = (float*)(ws + off); off += SZ_C;
    float* c1   = (float*)(ws + off); off += SZ_C;
    float* pred = (float*)(ws + off); off += 4096;
    size_t state_bytes = off - state_off;

    // ---- init: convert weights, zero all state ----
    {
        int n = G4 * H_DIM;
        f32_to_bf16_kernel<<<(n + 255) / 256, 256, 0, stream>>>(Whh0, Whh0b, n);
        int m = G4 * 1024;
        build_wcat_kernel<<<(m + 255) / 256, 256, 0, stream>>>(Wih1, Whh1, Wcat);
        zero_bytes_kernel<<<512, 256, 0, stream>>>((uint4*)(ws + state_off),
                                                   state_bytes / 16);
    }

    const dim3 grid(4, 32);      // 4 j-groups x 32 m-pair-tiles (32 rows each)
    const dim3 blk(256);
    const size_t shm0 = (size_t)32 * (H_DIM + 8) * 2 + 128;   // K=512
    const size_t shm1 = (size_t)32 * (1024  + 8) * 2 + 128;   // K=1024

    int p0 = 0, p1 = 0;   // current-state buffer index

    // ---- encode: T steps, layers fused per step ----
    for (int t = 0; t < T; ++t) {
        // Layer 0: gates = h0 @ Whh0^T + x[:,t] * Wih0 + b0
        lstm_step_kernel<H_DIM, true><<<grid, blk, shm0, stream>>>(
            h0b[p0], nullptr, Whh0b, b0,
            x + t, T, Wih0, c0, h0b[p0 ^ 1]);
        p0 ^= 1;
        // Layer 1: gates = [h0_t, h1_{t-1}] @ [Wih1;Whh1]^T + b1
        lstm_step_kernel<1024, false><<<grid, blk, shm1, stream>>>(
            h0b[p0], h1b[p1], Wcat, b1,
            nullptr, 0, nullptr, c1, h1b[p1 ^ 1]);
        p1 ^= 1;
    }

    // ---- decode: horizon autoregressive steps ----
    for (int t = 0; t < horizon; ++t) {
        head_kernel<<<B_DIM / 8, blk, 0, stream>>>(
            h1b[p1], headw, headb, pred, out, t, horizon);
        lstm_step_kernel<H_DIM, true><<<grid, blk, shm0, stream>>>(
            h0b[p0], nullptr, Whh0b, b0,
            pred, 1, Wih0, c0, h0b[p0 ^ 1]);
        p0 ^= 1;
        lstm_step_kernel<1024, false><<<grid, blk, shm1, stream>>>(
            h0b[p0], h1b[p1], Wcat, b1,
            nullptr, 0, nullptr, c1, h1b[p1 ^ 1]);
        p1 ^= 1;
    }
}
